// SplitEPA_39702677684944
// MI455X (gfx1250) — compile-verified, matrix operand order checked by reading, not run on hardware
//
#include <hip/hip_runtime.h>
#include <hip/hip_bf16.h>
#include <math.h>

// ---------------- CDNA5 WMMA types ----------------
typedef __attribute__((ext_vector_type(16))) _Float16 v16h;
typedef __attribute__((ext_vector_type(8)))  _Float16 v8h;
typedef __attribute__((ext_vector_type(8)))  float    v8f;

__device__ __forceinline__ v8f zero8() {
  v8f v;
#pragma unroll
  for (int i = 0; i < 8; ++i) v[i] = 0.0f;
  return v;
}

// Contiguous-K fragment load: two 16-byte vector loads per lane.
// CDNA5 ISA 7.12.2 layout: lanes 0-15 elems0-7=K kk..kk+7, elems8-15=K kk+16..23;
// lanes 16-31 shifted by 8. `base` points at this lane's row (A) / column (B).
__device__ __forceinline__ v16h load_frag(const _Float16* __restrict__ base,
                                          int kk, int lane) {
  const int o = kk + ((lane >> 4) << 3);
  const v8h lo = *(const v8h*)(base + o);
  const v8h hi = *(const v8h*)(base + o + 16);
  return __builtin_shufflevector(lo, hi, 0, 1, 2, 3, 4, 5, 6, 7,
                                 8, 9, 10, 11, 12, 13, 14, 15);
}

// ---------------- Generic contiguous-K f16 GEMM, f16 out ----------------
// A: row-major, row stride a_rs (K contiguous). B: col stride b_cs (K contiguous).
// Wave computes a 16 x (16*NSUB) strip (A fragment reused by NSUB WMMAs).
// NSUB is compile-time so the k-loop is straight-line (no exec masking).
// blockIdx.y/z are batch dims with independent strides.
template <int NSUB>
__global__ __launch_bounds__(256) void gemm_f16_kernel(
    const _Float16* __restrict__ A, long a_rs, long a_ybs, long a_zbs,
    const _Float16* __restrict__ Bm, long b_cs, long b_ybs, long b_zbs,
    _Float16* __restrict__ C, long c_rs, long c_cs, long c_ybs, long c_zbs,
    int M, int Ncols, int K) {
  const int lane = threadIdx.x & 31;
  const int wave = blockIdx.x * (blockDim.x >> 5) + (threadIdx.x >> 5);
  const int tn_cnt = Ncols / (16 * NSUB);
  const int tm = wave / tn_cnt;
  const int tn = wave - tm * tn_cnt;
  if (tm * 16 >= M) return;

  A  += (long)blockIdx.y * a_ybs + (long)blockIdx.z * a_zbs;
  Bm += (long)blockIdx.y * b_ybs + (long)blockIdx.z * b_zbs;
  C  += (long)blockIdx.y * c_ybs + (long)blockIdx.z * c_zbs;

  const int m0 = tm * 16;
  const int n0 = tn * 16 * NSUB;

  const _Float16* arow = A + (long)(m0 + (lane & 15)) * a_rs;
  const _Float16* bcol[NSUB];
#pragma unroll
  for (int s = 0; s < NSUB; ++s)
    bcol[s] = Bm + (long)(n0 + s * 16 + (lane & 15)) * b_cs;

  v8f acc[NSUB];
#pragma unroll
  for (int s = 0; s < NSUB; ++s) acc[s] = zero8();

  for (int kk = 0; kk < K; kk += 32) {
    const v16h af = load_frag(arow, kk, lane);
#pragma unroll
    for (int s = 0; s < NSUB; ++s) {
      const v16h bf = load_frag(bcol[s], kk, lane);
      acc[s] = __builtin_amdgcn_wmma_f32_16x16x32_f16(
          false, af, false, bf, (short)0, acc[s], false, false);
    }
  }

  const int rofs = (lane >> 4) << 3;  // lanes 16-31 hold M rows r+8
  if (c_rs == 1) {                    // 8 contiguous M per lane -> vector store
#pragma unroll
    for (int s = 0; s < NSUB; ++s) {
      const int n = n0 + s * 16 + (lane & 15);
      v8h h8;
#pragma unroll
      for (int r = 0; r < 8; ++r) h8[r] = (_Float16)acc[s][r];
      *(v8h*)(C + (long)n * c_cs + m0 + rofs) = h8;
    }
  } else {
#pragma unroll
    for (int s = 0; s < NSUB; ++s) {
      const int n = n0 + s * 16 + (lane & 15);
#pragma unroll
      for (int r = 0; r < 8; ++r)
        C[(long)(m0 + r + rofs) * c_rs + (long)n * c_cs] = (_Float16)acc[s][r];
    }
  }
}

// ---------------- QKVV projection with dual-layout epilogue ----------------
// QT[b][oc][n]  (vector stores, K-contiguous for all downstream reductions)
// VCA[b][n][e]  (row-major slice oc 256..383, K=e contiguous for x_ca)
__global__ __launch_bounds__(256) void gemm_g1_kernel(
    const _Float16* __restrict__ Xh,   // [65536][128]
    const _Float16* __restrict__ Wqh,  // [512][128]
    _Float16* __restrict__ QT,         // [4][512][16384]
    _Float16* __restrict__ VCA) {      // [4][16384][128]
  const int lane = threadIdx.x & 31;
  const int wave = blockIdx.x * (blockDim.x >> 5) + (threadIdx.x >> 5);
  const int tm = wave >> 3;       // 4096 row tiles
  const int tn = wave & 7;        // 8 col-of-64 tiles
  const int m0 = tm * 16;
  const int n0 = tn * 64;

  const _Float16* arow = Xh + (long)(m0 + (lane & 15)) * 128;
  const _Float16* bcol0 = Wqh + (long)(n0 + (lane & 15)) * 128;

  v8f acc[4] = {zero8(), zero8(), zero8(), zero8()};
#pragma unroll
  for (int kk = 0; kk < 128; kk += 32) {
    const v16h af = load_frag(arow, kk, lane);
#pragma unroll
    for (int s = 0; s < 4; ++s) {
      const v16h bf = load_frag(bcol0 + (long)s * 16 * 128, kk, lane);
      acc[s] = __builtin_amdgcn_wmma_f32_16x16x32_f16(
          false, af, false, bf, (short)0, acc[s], false, false);
    }
  }

  const int rofs = (lane >> 4) << 3;
  const int b = m0 >> 14;
  const int nl = (m0 & 16383) + rofs;  // local n of first of 8 contiguous rows
#pragma unroll
  for (int s = 0; s < 4; ++s) {
    const int col = n0 + s * 16 + (lane & 15);
    v8h h8;
#pragma unroll
    for (int r = 0; r < 8; ++r) h8[r] = (_Float16)acc[s][r];
    *(v8h*)(QT + (long)b * 8388608 + (long)col * 16384 + nl) = h8;
    if (col >= 256 && col < 384) {   // v_ca slice, row-major
      _Float16* p = VCA + ((long)b * 16384 + nl) * 128 + (col - 256);
#pragma unroll
      for (int r = 0; r < 8; ++r) p[(long)r * 128] = h8[r];
    }
  }
}

// ---------------- Split-K GEMM, f32 atomic-add out ----------------
// Huge-K / tiny-MN reductions (K=16384, out 32x64 / 32x32): each wave owns one
// 16x16 tile over one 2048-wide K chunk, reduced via global f32 atomics.
__global__ __launch_bounds__(256) void gemm_f16_splitk_kernel(
    const _Float16* __restrict__ A, long a_rs, long a_ybs, long a_zbs,
    const _Float16* __restrict__ Bm, long b_cs, long b_ybs, long b_zbs,
    float* __restrict__ C, long c_rs, long c_cs, long c_ybs, long c_zbs,
    int M, int Ncols, int KC) {
  const int lane = threadIdx.x & 31;
  const int wave = blockIdx.x * (blockDim.x >> 5) + (threadIdx.x >> 5);
  const int tiles_m = M >> 4, tiles_n = Ncols >> 4;
  const int tiles = tiles_m * tiles_n;
  const int chunk = wave / tiles;
  const int tile  = wave - chunk * tiles;
  const int tm = tile / tiles_n;
  const int tn = tile - tm * tiles_n;

  A  += (long)blockIdx.y * a_ybs + (long)blockIdx.z * a_zbs;
  Bm += (long)blockIdx.y * b_ybs + (long)blockIdx.z * b_zbs;
  C  += (long)blockIdx.y * c_ybs + (long)blockIdx.z * c_zbs;

  const int kbeg = chunk * KC;
  const _Float16* arow = A + (long)(tm * 16 + (lane & 15)) * a_rs;
  const _Float16* bcol = Bm + (long)(tn * 16 + (lane & 15)) * b_cs;

  v8f acc = zero8();
  for (int kk = kbeg; kk < kbeg + KC; kk += 32) {
    __builtin_prefetch(arow + kk + 128, 0, 1);
    __builtin_prefetch(bcol + kk + 128, 0, 1);
    const v16h af = load_frag(arow, kk, lane);
    const v16h bf = load_frag(bcol, kk, lane);
    acc = __builtin_amdgcn_wmma_f32_16x16x32_f16(
        false, af, false, bf, (short)0, acc, false, false);
  }
  const int rofs = (lane >> 4) << 3;
  const int n = tn * 16 + (lane & 15);
#pragma unroll
  for (int r = 0; r < 8; ++r)
    atomicAdd(C + (long)(tm * 16 + r + rofs) * c_rs + (long)n * c_cs, acc[r]);
}

// ---------------- Output projection + bias + 5D scatter ----------------
// y[b,n,col] = A[b,n,:] @ W^T + bias, scattered through the faithful-to-source
// reshape: n -> (d=n%16, w=(n/16)%32, h=n/512); out[b, cofs+col, dofs+d, w, h].
__global__ __launch_bounds__(256) void gemm_out_kernel(
    const _Float16* __restrict__ A,   // [65536,128] f16
    const _Float16* __restrict__ W,   // [64,128] f16
    const float* __restrict__ bias,   // [64]
    float* __restrict__ out,          // [4,128,32,32,32] f32
    int cofs, int dofs) {
  const int lane = threadIdx.x & 31;
  const int wave = blockIdx.x * (blockDim.x >> 5) + (threadIdx.x >> 5);
  const int m0 = wave * 16;

  const _Float16* arow = A + (long)(m0 + (lane & 15)) * 128;
  v8f acc[4] = {zero8(), zero8(), zero8(), zero8()};
#pragma unroll
  for (int kk = 0; kk < 128; kk += 32) {
    const v16h af = load_frag(arow, kk, lane);
#pragma unroll
    for (int s = 0; s < 4; ++s) {
      const v16h bf = load_frag(W + (long)(s * 16 + (lane & 15)) * 128, kk, lane);
      acc[s] = __builtin_amdgcn_wmma_f32_16x16x32_f16(
          false, af, false, bf, (short)0, acc[s], false, false);
    }
  }
  const int rofs = (lane >> 4) << 3;
#pragma unroll
  for (int s = 0; s < 4; ++s) {
    const int col = s * 16 + (lane & 15);
    const float bv = bias[col];
#pragma unroll
    for (int r = 0; r < 8; ++r) {
      const int m = m0 + r + rofs;
      const int b = m >> 14, n = m & 16383;
      const int d_ = n & 15, w_ = (n >> 4) & 31, h_ = n >> 9;
      out[(long)b * 4194304 + (long)(cofs + col) * 32768 +
          (long)(dofs + d_) * 1024 + w_ * 32 + h_] = acc[s][r] + bv;
    }
  }
}

// ---------------- Support kernels ----------------
// LDS-tiled gather/transpose of one depth-half of x into Xh[b][n][c] f16.
__global__ void gather_x_kernel(const float* __restrict__ x,
                                _Float16* __restrict__ Xh, int hf) {
  __shared__ _Float16 tile[32][72];
  const int b = blockIdx.z, dd = blockIdx.y;
  const int c0 = (blockIdx.x & 3) * 32;
  const int r0 = (blockIdx.x >> 2) * 64;
  const float* src = x + ((long)(b * 128) * 32 + hf * 16 + dd) * 1024;
#pragma unroll
  for (int p = 0; p < 8; ++p) {
    const int r = threadIdx.x & 63;
    const int ci = (threadIdx.x >> 6) + p * 4;
    tile[ci][r] = (_Float16)src[(long)(c0 + ci) * 32768 + r0 + r];
  }
  __syncthreads();
  _Float16* dst = Xh + ((long)b * 16384 + dd * 1024) * 128;
#pragma unroll
  for (int p = 0; p < 8; ++p) {
    const int c = threadIdx.x & 31;
    const int rr = (threadIdx.x >> 5) + p * 8;
    dst[(long)(r0 + rr) * 128 + c0 + c] = tile[c][rr];
  }
}

// vectorized f32 -> f16 (count % 8 == 0)
__global__ void cvt_f32_f16_kernel(const float* __restrict__ src,
                                   _Float16* __restrict__ dst, int count8) {
  const int i = blockIdx.x * blockDim.x + threadIdx.x;
  if (i >= count8) return;
  const long o = (long)i * 8;
  v8h h8;
#pragma unroll
  for (int j = 0; j < 8; ++j) h8[j] = (_Float16)src[o + j];
  *(v8h*)(dst + o) = h8;
}

// KP[bh][d][p] f32 -> KPhT[bh][p][d] f16 (K=d contiguous for SA-logits GEMM)
__global__ void cvt_trans_kp_kernel(const float* __restrict__ KP,
                                    _Float16* __restrict__ KPhT) {
  const int i = blockIdx.x * blockDim.x + threadIdx.x;  // < 32768
  const int p = i & 63, d = (i >> 6) & 31, bh = i >> 11;
  KPhT[(long)bh * 2048 + p * 32 + d] = (_Float16)KP[i];
}

// inv-L2-norm over contiguous rows of QT (vector loads).
__global__ void rownorm_kernel(const _Float16* __restrict__ QT, int cbase,
                               float* __restrict__ inv) {
  const int r = blockIdx.x;  // 0..511 : b=r>>7, hd=r&127
  const _Float16* p = QT + ((long)(r >> 7) * 512 + cbase + (r & 127)) * 16384;
  float s = 0.0f;
  for (int n = threadIdx.x * 8; n < 16384; n += blockDim.x * 8) {
    const v8h v = *(const v8h*)(p + n);
#pragma unroll
    for (int j = 0; j < 8; ++j) { const float f = (float)v[j]; s += f * f; }
  }
#pragma unroll
  for (int m = 16; m >= 1; m >>= 1) s += __shfl_xor(s, m, 32);
  __shared__ float red[8];
  if ((threadIdx.x & 31) == 0) red[threadIdx.x >> 5] = s;
  __syncthreads();
  if (threadIdx.x == 0) {
    float t = 0.0f;
#pragma unroll
    for (int i = 0; i < 8; ++i) t += red[i];
    inv[r] = 1.0f / fmaxf(sqrtf(t), 1e-12f);
  }
}

// qn/kn: scaled f16 rows in [b][h][d][n] (vector) + optional transposed copy.
__global__ void make_qn_kernel(const _Float16* __restrict__ QT, int cbase,
                               const float* __restrict__ inv,
                               _Float16* __restrict__ out,
                               _Float16* __restrict__ out_t) {
  const long i8 = ((long)blockIdx.x * blockDim.x + threadIdx.x) * 8;  // <8388608
  const int n = (int)(i8 & 16383);
  const int r = (int)(i8 >> 14);
  const float sc = inv[r];
  const v8h v = *(const v8h*)(QT + ((long)(r >> 7) * 512 + cbase + (r & 127)) * 16384 + n);
  v8h o;
#pragma unroll
  for (int j = 0; j < 8; ++j) o[j] = (_Float16)((float)v[j] * sc);
  *(v8h*)(out + i8) = o;
  if (out_t) {
    _Float16* t = out_t + (long)(r >> 5) * 524288 + (long)n * 32 + (r & 31);
#pragma unroll
    for (int j = 0; j < 8; ++j) t[(long)j * 32] = o[j];
  }
}

// softmax over rows of 32 (channel attention), one wave per row.
__global__ void softmax_ca_kernel(const float* __restrict__ CA,
                                  const float* __restrict__ t1,
                                  _Float16* __restrict__ out) {
  const int row = blockIdx.x;       // 512 rows [b][h][d]
  const int lane = threadIdx.x;     // 32
  const float v = CA[row * 32 + lane] * t1[(row >> 5) & 3];
  float m = v;
#pragma unroll
  for (int s = 16; s >= 1; s >>= 1) m = fmaxf(m, __shfl_xor(m, s, 32));
  const float e = __expf(v - m);
  float sum = e;
#pragma unroll
  for (int s = 16; s >= 1; s >>= 1) sum += __shfl_xor(sum, s, 32);
  out[row * 32 + lane] = (_Float16)(e / sum);
}

// softmax over rows of 64 (spatial attention), in place, one wave per row.
__global__ void softmax_sa_kernel(_Float16* __restrict__ SA,
                                  const float* __restrict__ t2) {
  const int row = blockIdx.x * 8 + (threadIdx.x >> 5);  // 262144 rows [bh][n]
  const int lane = threadIdx.x & 31;
  const float tt = t2[(row >> 14) & 3];
  const long base = (long)row * 64;
  const float v0 = (float)SA[base + lane] * tt;
  const float v1 = (float)SA[base + 32 + lane] * tt;
  float m = fmaxf(v0, v1);
#pragma unroll
  for (int s = 16; s >= 1; s >>= 1) m = fmaxf(m, __shfl_xor(m, s, 32));
  const float e0 = __expf(v0 - m), e1 = __expf(v1 - m);
  float sum = e0 + e1;
#pragma unroll
  for (int s = 16; s >= 1; s >>= 1) sum += __shfl_xor(sum, s, 32);
  SA[base + lane]      = (_Float16)(e0 / sum);
  SA[base + 32 + lane] = (_Float16)(e1 / sum);
}

// ---------------- Host orchestration ----------------
extern "C" void kernel_launch(void* const* d_in, const int* in_sizes, int n_in,
                              void* d_out, int out_size, void* d_ws, size_t ws_size,
                              hipStream_t stream) {
  (void)in_sizes; (void)n_in; (void)out_size; (void)ws_size;
  const float* x = (const float*)d_in[0];
  float* out = (float*)d_out;

  char* wsb = (char*)d_ws;
  size_t off = 0;
  auto take = [&](size_t bytes) -> void* {
    void* p = wsb + off;
    off += (bytes + 255) & ~(size_t)255;
    return p;
  };
  _Float16* Xh   = (_Float16*)take(8388608ull * 2);   // [4][16384][128]
  _Float16* Wqh  = (_Float16*)take(65536ull * 2);     // [512][128]
  _Float16* WEh  = (_Float16*)take(1048576ull * 2);   // [64][16384]
  _Float16* Wo1h = (_Float16*)take(8192ull * 2);
  _Float16* Wo2h = (_Float16*)take(8192ull * 2);
  _Float16* QT   = (_Float16*)take(33554432ull * 2);  // [4][512][16384]
  _Float16* VCA  = (_Float16*)take(8388608ull * 2);   // [4][16384][128]
  _Float16* QN   = (_Float16*)take(8388608ull * 2);   // [4][4][32][16384]
  _Float16* KN   = (_Float16*)take(8388608ull * 2);
  _Float16* QNT  = (_Float16*)take(8388608ull * 2);   // [4][4][16384][32]
  float*    INVQ = (float*)take(512ull * 4);
  float*    INVK = (float*)take(512ull * 4);
  float*    KP   = (float*)take(32768ull * 4);        // [4][4][32][64]
  float*    VP   = (float*)take(32768ull * 4);
  _Float16* KPhT = (_Float16*)take(32768ull * 2);     // [bh][64][32]
  _Float16* VPh  = (_Float16*)take(32768ull * 2);     // [bh][32][64]
  float*    CA   = (float*)take(16384ull * 4);        // [4][4][32][32]
  _Float16* CAh  = (_Float16*)take(16384ull * 2);
  _Float16* SA   = (_Float16*)take(16777216ull * 2);  // [16][16384][64]
  _Float16* XSA  = (_Float16*)take(8388608ull * 2);
  _Float16* XCA  = (_Float16*)take(8388608ull * 2);

  auto gemm = [&](const _Float16* A, long ars, long ay, long az,
                  const _Float16* Bm, long bcs, long by, long bz,
                  _Float16* Cc, long crs, long ccs, long cy, long cz,
                  int M, int Ncols, int K, int gy, int gz) {
    if (Ncols % 64 == 0) {
      const int waves = (M >> 4) * (Ncols >> 6);
      dim3 grid((waves + 7) / 8, gy, gz);
      gemm_f16_kernel<4><<<grid, 256, 0, stream>>>(
          A, ars, ay, az, Bm, bcs, by, bz, Cc, crs, ccs, cy, cz, M, Ncols, K);
    } else {
      const int waves = (M >> 4) * (Ncols >> 5);
      dim3 grid((waves + 7) / 8, gy, gz);
      gemm_f16_kernel<2><<<grid, 256, 0, stream>>>(
          A, ars, ay, az, Bm, bcs, by, bz, Cc, crs, ccs, cy, cz, M, Ncols, K);
    }
  };

  for (int hf = 0; hf < 2; ++hf) {
    const float* Wqkvv = (const float*)d_in[1 + 8 * hf];
    const float* WE    = (const float*)d_in[2 + 8 * hf];
    const float* Wo1   = (const float*)d_in[3 + 8 * hf];
    const float* bo1   = (const float*)d_in[4 + 8 * hf];
    const float* Wo2   = (const float*)d_in[5 + 8 * hf];
    const float* bo2   = (const float*)d_in[6 + 8 * hf];
    const float* t1    = (const float*)d_in[7 + 8 * hf];
    const float* t2    = (const float*)d_in[8 + 8 * hf];

    // stage inputs / weights to f16
    gather_x_kernel<<<dim3(64, 16, 4), 256, 0, stream>>>(x, Xh, hf);
    cvt_f32_f16_kernel<<<32, 256, 0, stream>>>(Wqkvv, Wqh, 8192);
    cvt_f32_f16_kernel<<<512, 256, 0, stream>>>(WE, WEh, 131072);
    cvt_f32_f16_kernel<<<4, 256, 0, stream>>>(Wo1, Wo1h, 1024);
    cvt_f32_f16_kernel<<<4, 256, 0, stream>>>(Wo2, Wo2h, 1024);

    // QKVV projection -> QT [b][oc][n] + VCA [b][n][e]
    gemm_g1_kernel<<<4096, 256, 0, stream>>>(Xh, Wqh, QT, VCA);

    // inverse L2 norms of q (oc 0..127) and k (oc 128..255)
    rownorm_kernel<<<512, 256, 0, stream>>>(QT, 0, INVQ);
    rownorm_kernel<<<512, 256, 0, stream>>>(QT, 128, INVK);
    make_qn_kernel<<<4096, 256, 0, stream>>>(QT, 0, INVQ, QN, QNT);
    make_qn_kernel<<<4096, 256, 0, stream>>>(QT, 128, INVK, KN, (\
        _Float16*)nullptr);

    // split-K reductions over N=16384 (f32 atomic accumulation)
    hipMemsetAsync(CA, 0, 16384ull * 4, stream);
    hipMemsetAsync(KP, 0, 32768ull * 4, stream);
    hipMemsetAsync(VP, 0, 32768ull * 4, stream);
    // CA[b,h,d,e] = qn[d,:]·kn[e,:]
    gemm_f16_splitk_kernel<<<dim3(4, 4, 4), 256, 0, stream>>>(
        QN, 16384, 2097152, 524288,
        KN, 16384, 2097152, 524288,
        CA, 32, 1, 4096, 1024,
        32, 32, 2048);
    // KP[b,h,d,p] = k[d,:]·WE[p,:]
    gemm_f16_splitk_kernel<<<dim3(8, 4, 4), 256, 0, stream>>>(
        QT + 2097152, 16384, 8388608, 524288,
        WEh, 16384, 0, 0,
        KP, 64, 1, 8192, 2048,
        32, 64, 2048);
    // VP[b,h,d,p] = v_sa[d,:]·WE[p,:]
    gemm_f16_splitk_kernel<<<dim3(8, 4, 4), 256, 0, stream>>>(
        QT + 6291456, 16384, 8388608, 524288,
        WEh, 16384, 0, 0,
        VP, 64, 1, 8192, 2048,
        32, 64, 2048);
    cvt_trans_kp_kernel<<<128, 256, 0, stream>>>(KP, KPhT);
    cvt_f32_f16_kernel<<<16, 256, 0, stream>>>(VP, VPh, 4096);

    // channel attention softmax (rows of 32)
    softmax_ca_kernel<<<512, 32, 0, stream>>>(CA, t1, CAh);

    // spatial logits: SA[bh,n,p] = qn^T[n,d] @ KPhT[p,d]^T  (M=16384,N=64,K=32)
    gemm(QNT, 32, 2097152, 524288,
         KPhT, 32, 8192, 2048,
         SA, 64, 1, 4194304, 1048576,
         16384, 64, 32, 4, 4);
    softmax_sa_kernel<<<32768, 256, 0, stream>>>(SA, t2);

    // x_ca[b,n,h*32+d] = attn_ca[d,e] @ v_ca[n,e]^T  (M=32,N=16384,K=32)
    gemm(CAh, 32, 4096, 1024,
         VCA, 128, 2097152, 32,
         XCA, 1, 128, 2097152, 32,
         32, 16384, 32, 4, 4);

    // x_sa (scrambled layout): C(m=n,col=d) -> XSA + d*65536 + h*16384 + n
    gemm(SA, 64, 4194304, 1048576,
         VPh, 64, 8192, 2048,
         XSA, 1, 65536, 2097152, 16384,
         16384, 32, 64, 4, 4);

    // output projections + bias + scatter into [4,128,32,32,32]
    gemm_out_kernel<<<512, 256, 0, stream>>>(XSA, Wo1h, bo1, out, 0, hf * 16);
    gemm_out_kernel<<<512, 256, 0, stream>>>(XCA, Wo2h, bo2, out, 64, hf * 16);
  }
}
